// GriffinRecBlock_80642305950376
// MI455X (gfx1250) — matrix operand order repaired and checked by reading, not compiled, not run on hardware
//
#include <hip/hip_runtime.h>
#include <hip/hip_bf16.h>
#include <stdint.h>

// Flip to 0 if the async-LDS mnemonics fail to assemble.
#ifndef USE_ASYNC_COPY
#define USE_ASYNC_COPY 1
#endif

// ---------------------------------------------------------------------------
// Types
// ---------------------------------------------------------------------------
typedef __bf16 bf16_t;
typedef __attribute__((ext_vector_type(16))) __bf16 bf16x16;
typedef __attribute__((ext_vector_type(8)))  float  f32x8;

struct __attribute__((aligned(16))) U4 { unsigned int a, b, c, d; };

union FragAB {
    bf16x16 v;
    U4      q[2];
};

__device__ inline bf16_t to_bf16(float f) {
    union { float f; unsigned u; } x; x.f = f;
    unsigned r = x.u + 0x7FFFu + ((x.u >> 16) & 1u);   // round-to-nearest-even
    union { unsigned short s; bf16_t b; } y;
    y.s = (unsigned short)(r >> 16);
    return y.b;
}

// ---------------------------------------------------------------------------
// Async global -> LDS copy (CDNA5 GLOBAL_LOAD_ASYNC_TO_LDS_B128, ASYNCcnt).
// Each lane copies 16 bytes from its global address to its LDS address.
// Generic LDS pointers carry the LDS byte offset in the low 32 bits.
// ---------------------------------------------------------------------------
__device__ __forceinline__ void copy16_async(void* lds, const void* g) {
#if USE_ASYNC_COPY
    unsigned loff = (unsigned)(uintptr_t)lds;
    asm volatile("global_load_async_to_lds_b128 %0, %1, off"
                 :: "v"(loff), "v"(g) : "memory");
#else
    *(U4*)lds = *(const U4*)g;
#endif
}
__device__ __forceinline__ void async_wait_le4() {
#if USE_ASYNC_COPY
    asm volatile("s_wait_asynccnt 0x4" ::: "memory");
#endif
}
__device__ __forceinline__ void async_wait_le6() {
#if USE_ASYNC_COPY
    asm volatile("s_wait_asynccnt 0x6" ::: "memory");
#endif
}
__device__ __forceinline__ void async_wait_0() {
#if USE_ASYNC_COPY
    asm volatile("s_wait_asynccnt 0x0" ::: "memory");
#endif
}

// ---------------------------------------------------------------------------
// Constants (match reference)
// ---------------------------------------------------------------------------
#define BATCH    2
#define SEQ      4096
#define DMODEL   2048
#define DREC     2048
#define FFN_DIM  8192
#define CHUNK    64
#define MTOK     (BATCH * SEQ)          // 8192 tokens

#define BM 128
#define BN 128
#define BK 32

// ---------------------------------------------------------------------------
// Weight convert + transpose: Wt[n][k] = (bf16) W[k][n]
// ---------------------------------------------------------------------------
__global__ void conv_transpose_kernel(const float* __restrict__ W,
                                      bf16_t* __restrict__ Wt,
                                      int K, int N) {
    size_t i = (size_t)blockIdx.x * 256 + threadIdx.x;     // over N*K
    size_t total = (size_t)K * N;
    if (i >= total) return;
    int n = (int)(i / K);
    int k = (int)(i % K);
    Wt[i] = to_bf16(W[(size_t)k * N + n]);
}

// ---------------------------------------------------------------------------
// RMSNorm (fp32 in) -> bf16 out. One block per row.
// ---------------------------------------------------------------------------
__global__ __launch_bounds__(256)
void rmsnorm_bf16_kernel(const float* __restrict__ x,
                         const float* __restrict__ w,
                         bf16_t* __restrict__ h, int D) {
    __shared__ float red[256];
    int row = blockIdx.x;
    const float* xr = x + (size_t)row * D;
    float ss = 0.f;
    for (int c = threadIdx.x; c < D; c += 256) { float v = xr[c]; ss += v * v; }
    red[threadIdx.x] = ss;
    __syncthreads();
    for (int s = 128; s > 0; s >>= 1) {
        if ((int)threadIdx.x < s) red[threadIdx.x] += red[threadIdx.x + s];
        __syncthreads();
    }
    float norm = rsqrtf(red[0] / (float)D + 1e-6f);
    bf16_t* hr = h + (size_t)row * D;
    for (int c = threadIdx.x; c < D; c += 256)
        hr[c] = to_bf16(xr[c] * norm * w[c]);
}

// ---------------------------------------------------------------------------
// Tiled WMMA GEMM:  out[M,N] = A[M,K] (bf16, row-major) @ Wt[N,K]^T (bf16)
// 256 threads = 8 waves; block tile 128x128; wave tile 32x64 (2x4 of 16x16).
// Double-buffered LDS; next tile streamed via async global->LDS copies that
// overlap the current tile's WMMAs. Steady-state loop is branch-free; the
// last K-step is peeled. KDIM is a compile-time constant.
// EPI: 0 raw f32, 1 sigmoid, 2 sigmoid(acc + bias[n]), 3 acc + resid[m,n]
// ---------------------------------------------------------------------------
template <int EPI, int KDIM>
__global__ __launch_bounds__(256)
void gemm_bf16_kernel(const bf16_t* __restrict__ A,
                      const bf16_t* __restrict__ Wt,
                      float* __restrict__ out,
                      const float* __restrict__ bias,
                      const float* __restrict__ resid,
                      int Ndim) {
    __shared__ bf16_t As[2][BM][BK];
    __shared__ bf16_t Bs[2][BN][BK];

    const int tid  = threadIdx.x;
    const int lane = tid & 31;
    const int wave = tid >> 5;
    const int wm   = wave & 3;          // 4 waves down
    const int wn   = wave >> 2;         // 2 waves across
    const int m0   = blockIdx.y * BM;
    const int n0   = blockIdx.x * BN;
    const int lrow = lane & 15;
    const int lhi  = lane >> 4;

    // Per-thread staging coordinates: 512 16B vectors per tile, 2 per thread.
    const int r0 = tid >> 2,           c0 = (tid & 3) * 8;
    const int r1 = (tid + 256) >> 2,   c1 = ((tid + 256) & 3) * 8;
    const bf16_t* gA0 = &A [(size_t)(m0 + r0) * KDIM + c0];
    const bf16_t* gA1 = &A [(size_t)(m0 + r1) * KDIM + c1];
    const bf16_t* gB0 = &Wt[(size_t)(n0 + r0) * KDIM + c0];
    const bf16_t* gB1 = &Wt[(size_t)(n0 + r1) * KDIM + c1];

    f32x8 acc[2][4];
    for (int i = 0; i < 2; ++i)
        for (int j = 0; j < 4; ++j)
            acc[i][j] = (f32x8)(0.0f);

    auto stage = [&](int buf, int k0) {
        copy16_async(&As[buf][r0][c0], gA0 + k0);
        copy16_async(&As[buf][r1][c1], gA1 + k0);
        copy16_async(&Bs[buf][r0][c0], gB0 + k0);
        copy16_async(&Bs[buf][r1][c1], gB1 + k0);
    };

    auto compute = [&](int buf) {
        FragAB af[2];
        for (int i = 0; i < 2; ++i) {
            int row = wm * 32 + i * 16 + lrow;
            int kb  = lhi * 16;                // A layout: contiguous K kb..kb+15
            af[i].q[0] = *(const U4*)&As[buf][row][kb];
            af[i].q[1] = *(const U4*)&As[buf][row][kb + 8];
        }
        for (int j = 0; j < 4; ++j) {
            FragAB bfg;
            int col = wn * 64 + j * 16 + lrow;
            int kb  = lhi * 16;                // B layout: contiguous K kb..kb+15
            bfg.q[0] = *(const U4*)&Bs[buf][col][kb];
            bfg.q[1] = *(const U4*)&Bs[buf][col][kb + 8];
            for (int i = 0; i < 2; ++i)
                acc[i][j] = __builtin_amdgcn_wmma_f32_16x16x32_bf16(
                    false, af[i].v, false, bfg.v, (short)0, acc[i][j],
                    false, false);
        }
    };

    constexpr int nk = KDIM / BK;
    stage(0, 0);
    for (int kt = 0; kt < nk - 1; ++kt) {      // branch-free steady state
        const int buf = kt & 1;
        stage(buf ^ 1, (kt + 1) * BK);
        async_wait_le4();                      // previous tile's group done
        __syncthreads();
        compute(buf);
        __syncthreads();                       // reads done before re-stage
    }
    async_wait_0();                            // peeled last K-step
    __syncthreads();
    compute((nk - 1) & 1);

    // Epilogue. C layout: lane%16 = N, element e -> M = e + 8*(lane/16).
    for (int i = 0; i < 2; ++i)
        for (int j = 0; j < 4; ++j) {
            int nn = n0 + wn * 64 + j * 16 + lrow;
            for (int e = 0; e < 8; ++e) {
                int   mm  = m0 + wm * 32 + i * 16 + lhi * 8 + e;
                float v   = acc[i][j][e];
                size_t idx = (size_t)mm * Ndim + nn;
                if (EPI == 1)       v = 1.f / (1.f + __expf(-v));
                else if (EPI == 2)  v = 1.f / (1.f + __expf(-(v + bias[nn])));
                else if (EPI == 3)  v = v + resid[idx];
                out[idx] = v;
            }
        }
}

// ---------------------------------------------------------------------------
// Fused gate/up WMMA GEMM + SiLU(gate)*up -> bf16 mid[M, FFN]
// Wt is [2*FFN, K]: rows [0,FFN) = gate weights, rows [FFN,2FFN) = up weights.
// Same double-buffered async-LDS pipeline (6 async ops per thread per tile).
// ---------------------------------------------------------------------------
__global__ __launch_bounds__(256)
void gemm_gateup_kernel(const bf16_t* __restrict__ A,
                        const bf16_t* __restrict__ Wt,
                        bf16_t* __restrict__ mid) {
    constexpr int KDIM = DMODEL;
    constexpr int FD   = FFN_DIM;
    __shared__ bf16_t As[2][BM][BK];
    __shared__ bf16_t Gs[2][BN][BK];
    __shared__ bf16_t Us[2][BN][BK];

    const int tid  = threadIdx.x;
    const int lane = tid & 31;
    const int wave = tid >> 5;
    const int wm   = wave & 3;
    const int wn   = wave >> 2;
    const int m0   = blockIdx.y * BM;
    const int n0   = blockIdx.x * BN;
    const int lrow = lane & 15;
    const int lhi  = lane >> 4;

    const int r0 = tid >> 2,          c0 = (tid & 3) * 8;
    const int r1 = (tid + 256) >> 2,  c1 = ((tid + 256) & 3) * 8;
    const bf16_t* gA0 = &A [(size_t)(m0 + r0) * KDIM + c0];
    const bf16_t* gA1 = &A [(size_t)(m0 + r1) * KDIM + c1];
    const bf16_t* gG0 = &Wt[(size_t)(n0 + r0) * KDIM + c0];
    const bf16_t* gG1 = &Wt[(size_t)(n0 + r1) * KDIM + c1];
    const bf16_t* gU0 = &Wt[(size_t)(FD + n0 + r0) * KDIM + c0];
    const bf16_t* gU1 = &Wt[(size_t)(FD + n0 + r1) * KDIM + c1];

    f32x8 accg[2][4], accu[2][4];
    for (int i = 0; i < 2; ++i)
        for (int j = 0; j < 4; ++j) {
            accg[i][j] = (f32x8)(0.0f);
            accu[i][j] = (f32x8)(0.0f);
        }

    auto stage = [&](int buf, int k0) {
        copy16_async(&As[buf][r0][c0], gA0 + k0);
        copy16_async(&As[buf][r1][c1], gA1 + k0);
        copy16_async(&Gs[buf][r0][c0], gG0 + k0);
        copy16_async(&Gs[buf][r1][c1], gG1 + k0);
        copy16_async(&Us[buf][r0][c0], gU0 + k0);
        copy16_async(&Us[buf][r1][c1], gU1 + k0);
    };

    auto compute = [&](int buf) {
        FragAB af[2];
        for (int i = 0; i < 2; ++i) {
            int row = wm * 32 + i * 16 + lrow;
            int kb  = lhi * 16;
            af[i].q[0] = *(const U4*)&As[buf][row][kb];
            af[i].q[1] = *(const U4*)&As[buf][row][kb + 8];
        }
        for (int j = 0; j < 4; ++j) {
            int col = wn * 64 + j * 16 + lrow;
            int kb  = lhi * 16;
            FragAB bg, bu;
            bg.q[0] = *(const U4*)&Gs[buf][col][kb];
            bg.q[1] = *(const U4*)&Gs[buf][col][kb + 8];
            bu.q[0] = *(const U4*)&Us[buf][col][kb];
            bu.q[1] = *(const U4*)&Us[buf][col][kb + 8];
            for (int i = 0; i < 2; ++i) {
                accg[i][j] = __builtin_amdgcn_wmma_f32_16x16x32_bf16(
                    false, af[i].v, false, bg.v, (short)0, accg[i][j], false, false);
                accu[i][j] = __builtin_amdgcn_wmma_f32_16x16x32_bf16(
                    false, af[i].v, false, bu.v, (short)0, accu[i][j], false, false);
            }
        }
    };

    constexpr int nk = KDIM / BK;
    stage(0, 0);
    for (int kt = 0; kt < nk - 1; ++kt) {
        const int buf = kt & 1;
        stage(buf ^ 1, (kt + 1) * BK);
        async_wait_le6();
        __syncthreads();
        compute(buf);
        __syncthreads();
    }
    async_wait_0();
    __syncthreads();
    compute((nk - 1) & 1);

    for (int i = 0; i < 2; ++i)
        for (int j = 0; j < 4; ++j) {
            int nn = n0 + wn * 64 + j * 16 + lrow;
            for (int e = 0; e < 8; ++e) {
                int   mm = m0 + wm * 32 + i * 16 + lhi * 8 + e;
                float g  = accg[i][j][e];
                float u  = accu[i][j][e];
                float s  = g * (1.f / (1.f + __expf(-g)));   // SiLU
                mid[(size_t)mm * FD + nn] = to_bf16(s * u);
            }
        }
}

// ---------------------------------------------------------------------------
// input_signal = sqrt(clip(1 - a*a, 1e-8)) * i_gate * v   (in-place on v)
// ---------------------------------------------------------------------------
__global__ void signal_kernel(const float* __restrict__ a,
                              const float* __restrict__ ig,
                              float* __restrict__ vsig, size_t total) {
    size_t i = (size_t)blockIdx.x * 256 + threadIdx.x;
    if (i >= total) return;
    float av = a[i];
    float g  = sqrtf(fmaxf(1.f - av * av, 1e-8f));
    vsig[i]  = g * ig[i] * vsig[i];
}

// ---------------------------------------------------------------------------
// Chunked linear scan  y_t = a_t*y_{t-1} + v_t   over S, per (b,d) column.
// ---------------------------------------------------------------------------
__global__ void scan_pass1_kernel(const float* __restrict__ a,
                                  float* __restrict__ sig,
                                  float* __restrict__ P,
                                  float* __restrict__ L, int D) {
    int gid = blockIdx.x * 256 + threadIdx.x;     // over B*nchunks*D
    int d   = gid % D;
    int bc  = gid / D;                            // b*nchunks + chunk
    size_t base = (size_t)bc * CHUNK * D + d;
    float state = 0.f, prod = 1.f;
    for (int t = 0; t < CHUNK; ++t) {
        size_t idx = base + (size_t)t * D;
        float av = a[idx];
        state = av * state + sig[idx];
        prod *= av;
        sig[idx] = state;                         // y_local
    }
    P[gid] = prod;
    L[gid] = state;
}

__global__ void scan_pass2_kernel(const float* __restrict__ P,
                                  const float* __restrict__ L,
                                  float* __restrict__ carry,
                                  int nch, int D) {
    int gid = blockIdx.x * 256 + threadIdx.x;     // over B*D
    int d = gid % D;
    int b = gid / D;
    float c0 = 0.f;
    for (int c = 0; c < nch; ++c) {
        size_t idx = (size_t)(b * nch + c) * D + d;
        carry[idx] = c0;
        c0 = L[idx] + P[idx] * c0;
    }
}

__global__ void scan_pass3_kernel(const float* __restrict__ a,
                                  const float* __restrict__ ylocal,
                                  const float* __restrict__ carry,
                                  bf16_t* __restrict__ rec, int D) {
    int gid = blockIdx.x * 256 + threadIdx.x;     // over B*nchunks*D
    int d   = gid % D;
    int bc  = gid / D;
    size_t base = (size_t)bc * CHUNK * D + d;
    float cin = carry[gid];
    float cum = 1.f;
    for (int t = 0; t < CHUNK; ++t) {
        size_t idx = base + (size_t)t * D;
        cum *= a[idx];
        rec[idx] = to_bf16(ylocal[idx] + cum * cin);
    }
}

// ---------------------------------------------------------------------------
// Host launcher
// ---------------------------------------------------------------------------
extern "C" void kernel_launch(void* const* d_in, const int* in_sizes, int n_in,
                              void* d_out, int out_size, void* d_ws, size_t ws_size,
                              hipStream_t stream) {
    (void)in_sizes; (void)n_in; (void)out_size; (void)ws_size;

    const float* x          = (const float*)d_in[0];
    const float* pre_norm_w = (const float*)d_in[1];
    const float* w_a        = (const float*)d_in[2];
    const float* w_i        = (const float*)d_in[3];
    const float* w_v        = (const float*)d_in[4];
    const float* w_o        = (const float*)d_in[5];
    const float* decay_bias = (const float*)d_in[6];
    const float* ffn_norm_w = (const float*)d_in[7];
    const float* w_gate_up  = (const float*)d_in[8];
    const float* w_down     = (const float*)d_in[9];
    float* out = (float*)d_out;

    const int M    = MTOK;
    const int nch  = SEQ / CHUNK;                 // 64

    // Workspace carve-up (all sizes multiples of 256 B).
    char* ws = (char*)d_ws;
    size_t off = 0;
    auto alloc = [&](size_t bytes) { void* p = ws + off; off += bytes; return p; };

    bf16_t* wa_t  = (bf16_t*)alloc((size_t)DMODEL * DREC * 2);
    bf16_t* wi_t  = (bf16_t*)alloc((size_t)DMODEL * DREC * 2);
    bf16_t* wv_t  = (bf16_t*)alloc((size_t)DMODEL * DREC * 2);
    bf16_t* wo_t  = (bf16_t*)alloc((size_t)DREC * DMODEL * 2);
    bf16_t* wgu_t = (bf16_t*)alloc((size_t)DMODEL * 2 * FFN_DIM * 2);
    bf16_t* wd_t  = (bf16_t*)alloc((size_t)FFN_DIM * DMODEL * 2);
    bf16_t* hA    = (bf16_t*)alloc((size_t)M * DMODEL * 2);      // reused as h2
    float*  aF    = (float*) alloc((size_t)M * DREC * 4);
    float*  igF   = (float*) alloc((size_t)M * DREC * 4);
    float*  sigF  = (float*) alloc((size_t)M * DREC * 4);        // v -> sig -> y_local
    bf16_t* recB  = (bf16_t*)alloc((size_t)M * DREC * 2);
    float*  x2F   = (float*) alloc((size_t)M * DMODEL * 4);
    float*  Pbuf  = (float*) alloc((size_t)BATCH * nch * DREC * 4);
    float*  Lbuf  = (float*) alloc((size_t)BATCH * nch * DREC * 4);
    float*  Cbuf  = (float*) alloc((size_t)BATCH * nch * DREC * 4);
    bf16_t* midB  = (bf16_t*)aF;   // alias: aF+igF (134 MB) free once scan done

    auto conv = [&](const float* W, bf16_t* Wt, int K, int N) {
        size_t tot = (size_t)K * N;
        conv_transpose_kernel<<<(unsigned)((tot + 255) / 256), 256, 0, stream>>>(W, Wt, K, N);
    };

    // 1) weight conversion (bf16, transposed)
    conv(w_a,       wa_t,  DMODEL, DREC);
    conv(w_i,       wi_t,  DMODEL, DREC);
    conv(w_v,       wv_t,  DMODEL, DREC);
    conv(w_o,       wo_t,  DREC,   DMODEL);
    conv(w_gate_up, wgu_t, DMODEL, 2 * FFN_DIM);
    conv(w_down,    wd_t,  FFN_DIM, DMODEL);

    // 2) pre-norm
    rmsnorm_bf16_kernel<<<M, 256, 0, stream>>>(x, pre_norm_w, hA, DMODEL);

    // 3) projections (WMMA)
    dim3 gproj(DREC / BN, M / BM);
    gemm_bf16_kernel<2, DMODEL><<<gproj, 256, 0, stream>>>(hA, wa_t, aF,  decay_bias, nullptr, DREC);
    gemm_bf16_kernel<1, DMODEL><<<gproj, 256, 0, stream>>>(hA, wi_t, igF, nullptr,    nullptr, DREC);
    gemm_bf16_kernel<0, DMODEL><<<gproj, 256, 0, stream>>>(hA, wv_t, sigF,nullptr,    nullptr, DREC);

    // 4) input signal + chunked scan
    size_t tot = (size_t)M * DREC;
    signal_kernel<<<(unsigned)((tot + 255) / 256), 256, 0, stream>>>(aF, igF, sigF, tot);
    int cthreads = BATCH * nch * DREC;
    scan_pass1_kernel<<<cthreads / 256, 256, 0, stream>>>(aF, sigF, Pbuf, Lbuf, DREC);
    scan_pass2_kernel<<<(BATCH * DREC) / 256, 256, 0, stream>>>(Pbuf, Lbuf, Cbuf, nch, DREC);
    scan_pass3_kernel<<<cthreads / 256, 256, 0, stream>>>(aF, sigF, Cbuf, recB, DREC);

    // 5) out-proj + residual
    dim3 gop(DMODEL / BN, M / BM);
    gemm_bf16_kernel<3, DREC><<<gop, 256, 0, stream>>>(recB, wo_t, x2F, nullptr, x, DMODEL);

    // 6) FFN norm (reuse hA as h2)
    rmsnorm_bf16_kernel<<<M, 256, 0, stream>>>(x2F, ffn_norm_w, hA, DMODEL);

    // 7) fused gate/up GEMM + SiLU*up
    dim3 ggu(FFN_DIM / BN, M / BM);
    gemm_gateup_kernel<<<ggu, 256, 0, stream>>>(hA, wgu_t, midB);

    // 8) down GEMM + residual -> final output (fp32)
    gemm_bf16_kernel<3, FFN_DIM><<<gop, 256, 0, stream>>>(midB, wd_t, out, nullptr, x2F, DMODEL);
}
